// SimpleObjectMatchingModule_90426241450573
// MI455X (gfx1250) — matrix operand order repaired and checked by reading, not compiled, not run on hardware
//
#include <hip/hip_runtime.h>
#include <hip/hip_bf16.h>

#define M_ROWS 8192
#define N_COLS 8192
#define D_FEAT 256
#define SINK_ITERS 100
#define INV_EPS 10.0f
#define LOG2E 1.4426950408889634f
#define ROW_CHUNKS 64                 // colpass row split
#define CHUNK_ROWS (M_ROWS / ROW_CHUNKS)

typedef __bf16 bf16;
typedef bf16 v16bf  __attribute__((ext_vector_type(16)));
typedef bf16 bf16x8 __attribute__((ext_vector_type(8)));
typedef bf16 bf16x2 __attribute__((ext_vector_type(2)));
typedef float v8f   __attribute__((ext_vector_type(8)));

// ---------------- f32 -> bf16 conversion ----------------
__global__ void cvt_bf16_kernel(const float* __restrict__ in, bf16* __restrict__ out, int n) {
    int i = blockIdx.x * blockDim.x + threadIdx.x;
    if (i < n) out[i] = (bf16)in[i];
}

// ---------------- v init to ones ----------------
__global__ void init_v_kernel(float* __restrict__ v, int n) {
    int i = blockIdx.x * blockDim.x + threadIdx.x;
    if (i < n) v[i] = 1.0f;
}

// ---------------- GEMM + exp: E = exp(10 * A B^T) ----------------
// Per wave: 16-row M tile x four 16-col N tiles, K step 32, software-pipelined
// one K-step ahead so global loads overlap the v_wmma_f32_16x16x32_bf16 issue.
__global__ __launch_bounds__(256)
void gemm_exp_kernel(const bf16* __restrict__ A, const bf16* __restrict__ B,
                     bf16* __restrict__ Ebf, float* __restrict__ Kf,
                     const float* __restrict__ zp)
{
    const int lane  = threadIdx.x;        // 0..31
    const int wave  = threadIdx.y;        // 0..7
    const int mbase = blockIdx.y * 128 + wave * 16;
    const int nbase = blockIdx.x * 64;
    const int half  = lane >> 4;
    const int l16   = lane & 15;

    union Frag { v16bf v; bf16x8 h[2]; };

    const bf16* arow = A + (size_t)(mbase + l16) * D_FEAT;
    const bf16* brow[4];
#pragma unroll
    for (int t = 0; t < 4; ++t)
        brow[t] = B + (size_t)(nbase + t * 16 + l16) * D_FEAT;

    auto loadA = [&](Frag& f, int k0) {
        // A 16x32: lanes 0-15 K {k0..k0+7, k0+16..23}; lanes 16-31 K {k0+8..15, k0+24..31}
        f.h[0] = *(const bf16x8*)(arow + k0 + half * 8);
        f.h[1] = *(const bf16x8*)(arow + k0 + 16 + half * 8);
    };
    auto loadB = [&](Frag& f, int t, int k0) {
        // B 32x16: lane = column; lanes 0-15 K {k0..k0+7, k0+16..23}; lanes 16-31 K {k0+8..15, k0+24..31}
        f.h[0] = *(const bf16x8*)(brow[t] + k0 + half * 8);
        f.h[1] = *(const bf16x8*)(brow[t] + k0 + 16 + half * 8);
    };

    v8f acc[4];
#pragma unroll
    for (int t = 0; t < 4; ++t) acc[t] = (v8f)0.0f;

    Frag a_cur, a_nxt, b_cur[4], b_nxt[4];
    loadA(a_cur, 0);
#pragma unroll
    for (int t = 0; t < 4; ++t) loadB(b_cur[t], t, 0);

#pragma unroll
    for (int k0 = 0; k0 < D_FEAT; k0 += 32) {
        const int kn = k0 + 32;
        if (kn < D_FEAT) {
            loadA(a_nxt, kn);
#pragma unroll
            for (int t = 0; t < 4; ++t) loadB(b_nxt[t], t, kn);
        }
#pragma unroll
        for (int t = 0; t < 4; ++t)
            acc[t] = __builtin_amdgcn_wmma_f32_16x16x32_bf16(
                false, a_cur.v, false, b_cur[t].v, (short)0, acc[t], false, false);
        a_cur = a_nxt;
#pragma unroll
        for (int t = 0; t < 4; ++t) b_cur[t] = b_nxt[t];
    }

    (void)zp;
    // C/D layout: lanes 0-15: VGPR r -> (M=r, N=lane); lanes 16-31: (M=r+8, N=lane-16)
#pragma unroll
    for (int t = 0; t < 4; ++t) {
        const int col = nbase + t * 16 + l16;
#pragma unroll
        for (int r = 0; r < 8; ++r) {
            const int row = mbase + r + half * 8;
            float e = __builtin_amdgcn_exp2f(acc[t][r] * (INV_EPS * LOG2E));
            Kf[(size_t)row * (N_COLS + 1) + col] = e;   // f32 master copy
            Ebf[(size_t)row * N_COLS + col] = (bf16)e;  // bf16 streaming copy
        }
    }
}

// ---------------- row pass: u = 1 / (E' v) ----------------
// blocks 0..1023: 8 waves, one row each; block 1024: border sum -> u[M]
// v staged into LDS via async global->LDS DMA (ASYNCcnt-tracked).
__global__ __launch_bounds__(256)
void rowpass_kernel(const bf16* __restrict__ E, const float* __restrict__ v,
                    float* __restrict__ u, const float* __restrict__ zp)
{
    const int b   = blockIdx.x;
    const int tid = threadIdx.x;
    const float c = __builtin_amdgcn_exp2f(zp[0] * (INV_EPS * LOG2E));

    if (b == M_ROWS / 8) {                 // border block: u[M] = 1/(c * sum(v))
        __shared__ float red[256];
        float s = 0.f;
        for (int k = tid; k <= N_COLS; k += 256) s += v[k];
        red[tid] = s; __syncthreads();
        for (int off = 128; off > 0; off >>= 1) {
            if (tid < off) red[tid] += red[tid + off];
            __syncthreads();
        }
        if (tid == 0) u[M_ROWS] = 1.0f / (c * red[0]);
        return;
    }

    __shared__ __align__(16) float vs[N_COLS];    // 32 KB of 320 KB WGP LDS
    // async global->LDS staging: 8 x b128 per thread
    for (int k = tid * 4; k < N_COLS; k += 256 * 4) {
        unsigned lds_addr = (unsigned)(unsigned long long)(const void*)(vs + k);
        const float* gaddr = v + k;
        asm volatile("global_load_async_to_lds_b128 %0, %1, off"
                     :: "v"(lds_addr), "v"(gaddr) : "memory");
    }
    asm volatile("s_wait_asynccnt 0x0" ::: "memory");
    __syncthreads();

    const int lane = tid & 31;
    const int wave = tid >> 5;
    const int row  = b * 8 + wave;
    const bf16* er = E + (size_t)row * N_COLS;

    float a0 = 0.f, a1 = 0.f;
    for (int it0 = 0; it0 < N_COLS / 256; it0 += 8) {
        __builtin_prefetch(er + (it0 + 8) * 256 + lane * 8, 0, 0);  // global_prefetch_b8
#pragma unroll
        for (int it = it0; it < it0 + 8; ++it) {
            const int j = it * 256 + lane * 8;
            bf16x8 e = *(const bf16x8*)(er + j);
            float4 va = *(const float4*)(vs + j);
            float4 vb = *(const float4*)(vs + j + 4);
            a0 += (float)e[0]*va.x + (float)e[1]*va.y + (float)e[2]*va.z + (float)e[3]*va.w;
            a1 += (float)e[4]*vb.x + (float)e[5]*vb.y + (float)e[6]*vb.z + (float)e[7]*vb.w;
        }
    }
    float acc = a0 + a1;
#pragma unroll
    for (int m = 16; m > 0; m >>= 1) acc += __shfl_xor(acc, m, 32);
    if (lane == 0) u[row] = 1.0f / (acc + c * v[N_COLS]);
}

// ---------------- col pass stage 1: partial column sums ----------------
// grid (16 col-blocks, 64 row-chunks) = 1024 blocks; coalesced row-major streaming.
__global__ __launch_bounds__(256)
void colpass_partial_kernel(const bf16* __restrict__ E, const float* __restrict__ u,
                            float* __restrict__ partial)
{
    const int cb  = blockIdx.x;           // 0..15
    const int rb  = blockIdx.y;           // 0..ROW_CHUNKS-1
    const int tid = threadIdx.x;
    const int j0  = cb * 512 + tid * 2;
    const int i0  = rb * CHUNK_ROWS;

    const bf16* ep = E + (size_t)i0 * N_COLS + j0;
    float a0 = 0.f, a1 = 0.f;
    for (int ii = 0; ii < CHUNK_ROWS; ii += 32) {
        __builtin_prefetch(ep + (size_t)(ii + 32) * N_COLS, 0, 0);  // global_prefetch_b8
#pragma unroll 8
        for (int i = ii; i < ii + 32; ++i) {
            bf16x2 e = *(const bf16x2*)(ep + (size_t)i * N_COLS);
            float ui = u[i0 + i];                 // uniform -> scalar load
            a0 += (float)e[0] * ui;
            a1 += (float)e[1] * ui;
        }
    }
    partial[(size_t)rb * N_COLS + j0]     = a0;
    partial[(size_t)rb * N_COLS + j0 + 1] = a1;
}

// ---------------- col pass stage 2: reduce partials -> v ----------------
// blocks 0..31: v[j] = 1/(sum_r partial[r][j] + c*u[M]); block 32: border sum -> v[N]
__global__ __launch_bounds__(256)
void colpass_reduce_kernel(const float* __restrict__ partial, const float* __restrict__ u,
                           float* __restrict__ v, const float* __restrict__ zp)
{
    const int b   = blockIdx.x;
    const int tid = threadIdx.x;
    const float c = __builtin_amdgcn_exp2f(zp[0] * (INV_EPS * LOG2E));

    if (b == N_COLS / 256) {               // border block: v[N] = 1/(c * sum(u))
        __shared__ float red[256];
        float s = 0.f;
        for (int k = tid; k <= M_ROWS; k += 256) s += u[k];
        red[tid] = s; __syncthreads();
        for (int off = 128; off > 0; off >>= 1) {
            if (tid < off) red[tid] += red[tid + off];
            __syncthreads();
        }
        if (tid == 0) v[N_COLS] = 1.0f / (c * red[0]);
        return;
    }

    const int j = b * 256 + tid;
    float a = 0.f;
#pragma unroll 8
    for (int r = 0; r < ROW_CHUNKS; ++r)
        a += partial[(size_t)r * N_COLS + j];
    v[j] = 1.0f / (a + c * u[M_ROWS]);
}

// ---------------- finalize: K = diag(u) E' diag(v); P = K[:M,:N] ----------------
__global__ __launch_bounds__(256)
void finalize_kernel(const float* __restrict__ u, const float* __restrict__ v,
                     float* __restrict__ P, float* __restrict__ K,
                     const float* __restrict__ zp)
{
    const int j = blockIdx.x * 256 + threadIdx.x;
    const int i = blockIdx.y;
    if (j > N_COLS) return;
    const float c = __builtin_amdgcn_exp2f(zp[0] * (INV_EPS * LOG2E));
    const size_t kidx = (size_t)i * (N_COLS + 1) + j;
    const bool interior = (i < M_ROWS) && (j < N_COLS);
    const float e = interior ? K[kidx] : c;
    const float val = u[i] * e * v[j];
    K[kidx] = val;
    if (interior) P[(size_t)i * N_COLS + j] = val;
}

// ---------------- launcher ----------------
extern "C" void kernel_launch(void* const* d_in, const int* in_sizes, int n_in,
                              void* d_out, int out_size, void* d_ws, size_t ws_size,
                              hipStream_t stream) {
    const float* Mq = (const float*)d_in[0];   // 8192x256 f32
    const float* Nr = (const float*)d_in[1];   // 8192x256 f32
    const float* zp = (const float*)d_in[2];   // scalar z

    float* P = (float*)d_out;                               // 8192*8192 f32
    float* K = (float*)d_out + (size_t)M_ROWS * N_COLS;     // 8193*8193 f32

    // workspace layout
    bf16*  Ebf     = (bf16*)d_ws;                             // 8192*8192 bf16 (134 MB)
    bf16*  Abf     = Ebf + (size_t)M_ROWS * N_COLS;           // 8192*256 bf16
    bf16*  Bbf     = Abf + (size_t)M_ROWS * D_FEAT;           // 8192*256 bf16
    float* u       = (float*)(Bbf + (size_t)N_COLS * D_FEAT); // 8193 f32 (pad to 8704)
    float* v       = u + 8704;                                // keep 16B alignment
    float* partial = v + 8704;                                // 64*8192 f32 (2 MB)

    (void)in_sizes; (void)n_in; (void)out_size; (void)ws_size;

    // 1) convert inputs to bf16
    cvt_bf16_kernel<<<(M_ROWS * D_FEAT) / 256, 256, 0, stream>>>(Mq, Abf, M_ROWS * D_FEAT);
    cvt_bf16_kernel<<<(N_COLS * D_FEAT) / 256, 256, 0, stream>>>(Nr, Bbf, N_COLS * D_FEAT);

    // 2) v = 1
    init_v_kernel<<<(N_COLS + 1 + 255) / 256, 256, 0, stream>>>(v, N_COLS + 1);

    // 3) E = exp(S/eps) via WMMA bf16 GEMM (f32 copy in K interior, bf16 copy in ws)
    gemm_exp_kernel<<<dim3(N_COLS / 64, M_ROWS / 128), dim3(32, 8), 0, stream>>>(
        Abf, Bbf, Ebf, K, zp);

    // 4) 100 Sinkhorn iterations as scaling-vector mat-vecs over bf16 E
    for (int t = 0; t < SINK_ITERS; ++t) {
        rowpass_kernel<<<M_ROWS / 8 + 1, 256, 0, stream>>>(Ebf, v, u, zp);
        colpass_partial_kernel<<<dim3(N_COLS / 512, ROW_CHUNKS), 256, 0, stream>>>(
            Ebf, u, partial);
        colpass_reduce_kernel<<<N_COLS / 256 + 1, 256, 0, stream>>>(partial, u, v, zp);
    }

    // 5) scale out final K and P
    finalize_kernel<<<dim3((N_COLS + 1 + 255) / 256, M_ROWS + 1), 256, 0, stream>>>(
        u, v, P, K, zp);
}
